// Autorec_DG_13116830122688
// MI455X (gfx1250) — matrix operand order repaired and checked by reading, not compiled, not run on hardware
//
#include <hip/hip_runtime.h>
#include <hip/hip_bf16.h>

#define NU   20000    // users (rows)
#define NI   6000     // items
#define HID  500      // hidden
#define HP   512      // hidden padded to /16
#define KP1  6016     // NI padded to /32 (encoder weight row stride, bf16)
#define NIP  6016     // NI padded to /128 (decoder weight rows)
#define NE   160000   // edges
#define KMAIN (NI & ~31)   // 5984: full 32-wide K tiles in encoder GEMM

typedef __attribute__((ext_vector_type(16))) __bf16       bf16x16;
typedef __attribute__((ext_vector_type(8)))  float        f32x8;
typedef __attribute__((ext_vector_type(4)))  float        f32x4;
typedef __attribute__((ext_vector_type(4)))  unsigned int u32x4;
typedef __attribute__((ext_vector_type(2)))  unsigned int u32x2;

// LDS row stride (elements) for a 32-wide K tile: 40*2B = 80B rows keeps every
// fragment read 16B aligned (ds_load_b128) and banks distinct (20*row mod 64).
#define LDSS 40

// ---------------- CDNA5 async global->LDS (ASYNCcnt) ----------------
__device__ __forceinline__ void async_b128(unsigned lds_off, const void* gaddr) {
  asm volatile("global_load_async_to_lds_b128 %0, %1, off"
               :: "v"(lds_off), "v"((unsigned long long)(uintptr_t)gaddr)
               : "memory");
}
__device__ __forceinline__ void wait_async0() {
  asm volatile("s_wait_asynccnt 0" ::: "memory");
}
__device__ __forceinline__ unsigned lds_off32(const void* p) {
  return (unsigned)(uintptr_t)p;   // addrspace(3)->flat keeps LDS offset in low 32b
}

// ---------------- WMMA fragment loaders (ISA 7.12.2, wave32) ----------------
__device__ __forceinline__ bf16x16 frag_a(const __bf16* s, int lane, int row0) {
  const int m  = lane & 15;
  const int kb = (lane >> 4) << 3;               // 0 or 8
  const __bf16* p = s + (row0 + m) * LDSS;
  union { u32x4 q[2]; bf16x16 v; } u;
  u.q[0] = *(const u32x4*)(p + kb);              // K kb..kb+7
  u.q[1] = *(const u32x4*)(p + kb + 16);         // K kb+16..kb+23
  return u.v;
}
__device__ __forceinline__ bf16x16 frag_b(const __bf16* s, int lane, int row0) {
  const int n  = lane & 15;
  const int ks = (lane >> 4) << 3;               // 0 or 8
  const __bf16* p = s + (row0 + n) * LDSS + ks;
  union { u32x4 q[2]; bf16x16 v; } u;
  u.q[0] = *(const u32x4*)(p);
  u.q[1] = *(const u32x4*)(p + 16);
  return u.v;
}

// one 128x128x32 block-tile MMA step (8 waves, each 32x64)
__device__ __forceinline__ void mma_step(const __bf16* sA, const __bf16* sB,
                                         int lane, int wm, int wn,
                                         f32x8 (&acc)[2][4]) {
  bf16x16 af[2], bf[4];
  #pragma unroll
  for (int mt = 0; mt < 2; ++mt) af[mt] = frag_a(sA, lane, wm * 32 + mt * 16);
  #pragma unroll
  for (int nt = 0; nt < 4; ++nt) bf[nt] = frag_b(sB, lane, wn * 64 + nt * 16);
  #pragma unroll
  for (int mt = 0; mt < 2; ++mt)
    #pragma unroll
    for (int nt = 0; nt < 4; ++nt)
      acc[mt][nt] = __builtin_amdgcn_wmma_f32_16x16x32_bf16(
          false, af[mt], false, bf[nt], (short)0, acc[mt][nt], false, false);
}

// ---------------- workspace zero ----------------
__global__ __launch_bounds__(256) void zero_f32x4(f32x4* p, long n4) {
  long i = (long)blockIdx.x * blockDim.x + threadIdx.x;
  if (i < n4) p[i] = (f32x4){0.f, 0.f, 0.f, 0.f};
}

// ------------- weight pre-pack: f32 -> zero-padded bf16 ---------------------
__global__ __launch_bounds__(256)
void pack_encw(const float* __restrict__ w, __bf16* __restrict__ wb) {
  int i  = blockIdx.x * 256 + threadIdx.x;       // over 512 * (6016/4) groups
  int n  = i / (KP1 / 4);
  int k4 = (i - n * (KP1 / 4)) * 4;
  f32x4 v = (n < HID && k4 + 4 <= NI) ? *(const f32x4*)(w + (long)n * NI + k4)
                                      : (f32x4){0.f, 0.f, 0.f, 0.f};
  union { __bf16 b[4]; u32x2 d; } pk;
  pk.b[0] = (__bf16)v.x; pk.b[1] = (__bf16)v.y;
  pk.b[2] = (__bf16)v.z; pk.b[3] = (__bf16)v.w;
  *(u32x2*)(wb + (long)n * KP1 + k4) = pk.d;
}
__global__ __launch_bounds__(256)
void pack_decw(const float* __restrict__ w, __bf16* __restrict__ wb) {
  int i  = blockIdx.x * 256 + threadIdx.x;       // over 6016 * (512/4) groups
  int n  = i >> 7;                               // 128 groups per row
  int k4 = (i & 127) * 4;
  f32x4 v = (n < NI && k4 + 4 <= HID) ? *(const f32x4*)(w + (long)n * HID + k4)
                                      : (f32x4){0.f, 0.f, 0.f, 0.f};
  union { __bf16 b[4]; u32x2 d; } pk;
  pk.b[0] = (__bf16)v.x; pk.b[1] = (__bf16)v.y;
  pk.b[2] = (__bf16)v.z; pk.b[3] = (__bf16)v.w;
  *(u32x2*)(wb + (long)n * HP + k4) = pk.d;
}

// ---------------- GEMM1: h = sigmoid(x @ enc_w^T + enc_b), bf16 out --------
__global__ __launch_bounds__(256)
void enc_gemm(const float* __restrict__ x, const __bf16* __restrict__ wb,
              const float* __restrict__ bias, __bf16* __restrict__ h)
{
  __shared__ __attribute__((aligned(16))) __bf16 sA[128 * LDSS];
  __shared__ __attribute__((aligned(16))) __bf16 sB[128 * LDSS];
  const int t    = threadIdx.x;
  const int lane = t & 31;
  const int wv   = t >> 5;
  const int wm   = wv & 3;            // 4 M strips of 32 rows
  const int wn   = wv >> 2;           // 2 N strips of 64 cols
  const int bm   = blockIdx.y * 128;
  const int bn   = blockIdx.x * 128;

  f32x8 acc[2][4] = {};

  // ---- main loop: 187 full K tiles, completely guard-free staging ----
  for (int k0 = 0; k0 < KMAIN; k0 += 32) {
    #pragma unroll
    for (int i = 0; i < 4; ++i) {                // A: 128x32 f32 -> bf16
      int g = t + i * 256;                       // 0..1023
      int r = g >> 3, c4 = g & 7;
      int gm = bm + r; if (gm > NU - 1) gm = NU - 1;
      const float* gp = x + (long)gm * NI + k0 + c4 * 4;
      f32x4 v = *(const f32x4*)gp;
      __builtin_prefetch(gp + 32, 0, 1);         // next K tile (speculative)
      union { __bf16 b[4]; u32x2 d; } pk;
      pk.b[0] = (__bf16)v.x; pk.b[1] = (__bf16)v.y;
      pk.b[2] = (__bf16)v.z; pk.b[3] = (__bf16)v.w;
      *(u32x2*)(sA + r * LDSS + c4 * 4) = pk.d;
    }
    #pragma unroll
    for (int i = 0; i < 2; ++i) {                // B: async 128x32 bf16
      int c = t + i * 256;                       // 512 chunks of 16B
      int r = c >> 2, q = c & 3;
      async_b128(lds_off32(sB + r * LDSS + q * 8),
                 wb + (long)(bn + r) * KP1 + k0 + q * 8);
    }
    wait_async0();
    __syncthreads();
    mma_step(sA, sB, lane, wm, wn, acc);
    __syncthreads();
  }

  // ---- peeled K tail (k = 5984..5999 valid, 6000..6015 zero) ----
  {
    #pragma unroll
    for (int i = 0; i < 4; ++i) {
      int g = t + i * 256;
      int r = g >> 3, c4 = g & 7;
      int gm = bm + r; if (gm > NU - 1) gm = NU - 1;
      f32x4 v = (c4 < 4) ? *(const f32x4*)(x + (long)gm * NI + KMAIN + c4 * 4)
                         : (f32x4){0.f, 0.f, 0.f, 0.f};
      union { __bf16 b[4]; u32x2 d; } pk;
      pk.b[0] = (__bf16)v.x; pk.b[1] = (__bf16)v.y;
      pk.b[2] = (__bf16)v.z; pk.b[3] = (__bf16)v.w;
      *(u32x2*)(sA + r * LDSS + c4 * 4) = pk.d;
    }
    #pragma unroll
    for (int i = 0; i < 2; ++i) {                // padded weights: no guard
      int c = t + i * 256;
      int r = c >> 2, q = c & 3;
      async_b128(lds_off32(sB + r * LDSS + q * 8),
                 wb + (long)(bn + r) * KP1 + KMAIN + q * 8);
    }
    wait_async0();
    __syncthreads();
    mma_step(sA, sB, lane, wm, wn, acc);
  }

  // ---- epilogue: +bias, sigmoid, bf16 store into h[NU][HP]; pad cols -> 0 --
  const int cn = lane & 15;
  const int mo = (lane >> 4) * 8;
  #pragma unroll
  for (int mt = 0; mt < 2; ++mt)
    #pragma unroll
    for (int nt = 0; nt < 4; ++nt) {
      int col  = bn + wn * 64 + nt * 16 + cn;    // < 512 always
      float bc = (col < HID) ? bias[col] : 0.f;
      #pragma unroll
      for (int v = 0; v < 8; ++v) {
        int row = bm + wm * 32 + mt * 16 + mo + v;
        if (row < NU) {
          float s  = acc[mt][nt][v] + bc;
          float hv = (col < HID) ? (1.f / (1.f + __expf(-s))) : 0.f;
          h[(long)row * HP + col] = (__bf16)hv;
        }
      }
    }
}

// ---------------- edge gather-multiply-scatter (f32 atomics) ----------------
__global__ __launch_bounds__(256)
void graph_scatter(const int* __restrict__ ei, const float* __restrict__ ew,
                   const __bf16* __restrict__ h, float* __restrict__ agg)
{
  long idx = (long)blockIdx.x * blockDim.x + threadIdx.x;  // over NE*HP
  if (idx >= (long)NE * HP) return;
  int e = (int)(idx >> 9);
  int f = (int)(idx & (HP - 1));
  if (f >= HID) return;
  int s = ei[e];
  int d = ei[NE + e];
  float m = (float)h[(long)s * HP + f] * ew[e];
  atomicAdd(agg + (long)d * HP + f, m);
}

// ---- GEMM2: p = clip(mix @ dec_w^T + dec_b, 1, 5) with row/col masks ------
__global__ __launch_bounds__(256)
void dec_gemm(const float* __restrict__ agg, const __bf16* __restrict__ h,
              const __bf16* __restrict__ wb, const float* __restrict__ bias,
              const float* __restrict__ convw, const int* __restrict__ ft0,
              const int* __restrict__ ft1, const float* __restrict__ fillp,
              float* __restrict__ out)
{
  __shared__ __attribute__((aligned(16))) __bf16 sA[128 * LDSS];
  __shared__ __attribute__((aligned(16))) __bf16 sB[128 * LDSS];
  const int t    = threadIdx.x;
  const int lane = t & 31;
  const int wv   = t >> 5;
  const int wm   = wv & 3;
  const int wn   = wv >> 2;
  const int bm   = blockIdx.y * 128;
  const int bn   = blockIdx.x * 128;
  const float cw   = convw[0];
  const float icw  = 1.f - cw;
  const float fill = fillp[0];

  f32x8 acc[2][4] = {};

  for (int k0 = 0; k0 < HP; k0 += 32) {          // K = 512, no tail
    // ---- stage A = cw*agg + (1-cw)*h, f32 -> bf16 (guard-free) ----
    #pragma unroll
    for (int i = 0; i < 4; ++i) {
      int g = t + i * 256;
      int r = g >> 3, c4 = g & 7;
      int gm = bm + r; if (gm > NU - 1) gm = NU - 1;
      int gk = k0 + c4 * 4;
      f32x4 av = *(const f32x4*)(agg + (long)gm * HP + gk);
      union { u32x2 d; __bf16 b[4]; } hu;
      hu.d = *(const u32x2*)(h + (long)gm * HP + gk);
      union { __bf16 b[4]; u32x2 d; } pk;
      pk.b[0] = (__bf16)(cw * av.x + icw * (float)hu.b[0]);
      pk.b[1] = (__bf16)(cw * av.y + icw * (float)hu.b[1]);
      pk.b[2] = (__bf16)(cw * av.z + icw * (float)hu.b[2]);
      pk.b[3] = (__bf16)(cw * av.w + icw * (float)hu.b[3]);
      *(u32x2*)(sA + r * LDSS + c4 * 4) = pk.d;
    }
    // ---- stage B tile: async copy 128x32 bf16 from padded dec_wb ----
    #pragma unroll
    for (int i = 0; i < 2; ++i) {
      int c = t + i * 256;
      int r = c >> 2, q = c & 3;
      async_b128(lds_off32(sB + r * LDSS + q * 8),
                 wb + (long)(bn + r) * HP + k0 + q * 8);
    }
    wait_async0();
    __syncthreads();
    mma_step(sA, sB, lane, wm, wn, acc);
    __syncthreads();
  }

  // ---- epilogue: +bias, hardtanh(1,5), masks, f32 store ----
  const int cn = lane & 15;
  const int mo = (lane >> 4) * 8;
  #pragma unroll
  for (int mt = 0; mt < 2; ++mt)
    #pragma unroll
    for (int nt = 0; nt < 4; ++nt) {
      int col = bn + wn * 64 + nt * 16 + cn;
      if (col < NI) {
        float bc    = bias[col];
        int   cmask = ft1[col];
        #pragma unroll
        for (int v = 0; v < 8; ++v) {
          int row = bm + wm * 32 + mt * 16 + mo + v;
          if (row < NU) {
            float p = acc[mt][nt][v] + bc;
            p = fminf(fmaxf(p, 1.f), 5.f);
            if (ft0[row] == 0) p = fill;
            if (cmask == 0)    p = fill;
            out[(long)row * NI + col] = p;
          }
        }
      }
    }
}

extern "C" void kernel_launch(void* const* d_in, const int* in_sizes, int n_in,
                              void* d_out, int out_size, void* d_ws, size_t ws_size,
                              hipStream_t stream) {
  const float* x      = (const float*)d_in[0];
  const int*   ei     = (const int*)  d_in[1];
  const float* ew     = (const float*)d_in[2];
  const int*   ft0    = (const int*)  d_in[3];
  const int*   ft1    = (const int*)  d_in[4];
  const float* fill   = (const float*)d_in[5];
  const float* enc_w  = (const float*)d_in[6];
  const float* enc_b  = (const float*)d_in[7];
  const float* dec_w  = (const float*)d_in[8];
  const float* dec_b  = (const float*)d_in[9];
  const float* conv_w = (const float*)d_in[10];
  float* out = (float*)d_out;

  // workspace layout: h bf16 | agg f32 | enc_wb bf16 | dec_wb bf16
  char* ws = (char*)d_ws;
  __bf16* h      = (__bf16*)ws;                         ws += (size_t)NU * HP * 2;
  float*  agg    = (float*)ws;                          ws += (size_t)NU * HP * 4;
  __bf16* enc_wb = (__bf16*)ws;                         ws += (size_t)HP * KP1 * 2;
  __bf16* dec_wb = (__bf16*)ws;

  // 1) pre-pack weights into zero-padded bf16 (enables guard-free async B tiles)
  pack_encw<<<(HP * (KP1 / 4)) / 256, 256, 0, stream>>>(enc_w, enc_wb);
  pack_decw<<<(NIP * (HP / 4)) / 256, 256, 0, stream>>>(dec_w, dec_wb);

  // 2) zero agg
  long n4 = (long)NU * HP / 4;
  zero_f32x4<<<(int)((n4 + 255) / 256), 256, 0, stream>>>((f32x4*)agg, n4);

  // 3) encoder GEMM + sigmoid  (grid: N tiles x M tiles)
  dim3 g1(HP / 128, (NU + 127) / 128);
  enc_gemm<<<g1, 256, 0, stream>>>(x, enc_wb, enc_b, h);

  // 4) edge gather/scatter
  long ew_work = (long)NE * HP;
  graph_scatter<<<(int)((ew_work + 255) / 256), 256, 0, stream>>>(ei, ew, h, agg);

  // 5) decoder GEMM + epilogue
  dim3 g2((NI + 127) / 128, (NU + 127) / 128);
  dec_gemm<<<g2, 256, 0, stream>>>(agg, h, dec_wb, dec_b, conv_w, ft0, ft1, fill, out);
}
